// MMoE_11690900979996
// MI455X (gfx1250) — matrix operand order repaired and checked by reading, not compiled, not run on hardware
//
#include <hip/hip_runtime.h>
#include <stdint.h>

// ---------------- problem sizes (compile-time) ----------------
#define B_   16384
#define D_   1024
#define E_   8
#define T_   2
#define H1_  1024
#define H2_  512
#define DO_  256
#define TH1_ 512
#define TH2_ 256

// ---------------- types ----------------
typedef __bf16        v16bf __attribute__((ext_vector_type(16)));
typedef float         v8f   __attribute__((ext_vector_type(8)));
typedef unsigned int  v4u   __attribute__((ext_vector_type(4)));

__device__ __forceinline__ float bf2f(uint16_t h) {
  union { uint32_t u; float f; } v; v.u = ((uint32_t)h) << 16; return v.f;
}
__device__ __forceinline__ uint16_t f2bf(float f) {
  union { float f; uint32_t u; } v; v.f = f;
  uint32_t u = v.u;
  return (uint16_t)((u + 0x7FFFu + ((u >> 16) & 1u)) >> 16);  // RNE
}

// A-fragment: 8 contiguous bf16 at p, 8 more at p+16 (k, k+16 pattern)
__device__ __forceinline__ v16bf load_frag_split(const uint16_t* p) {
  union { v4u q[2]; v16bf v; } u;
  u.q[0] = *reinterpret_cast<const v4u*>(p);
  u.q[1] = *reinterpret_cast<const v4u*>(p + 16);
  return u.v;
}
// 16 contiguous bf16 at p (unused; B fragments use the split load)
__device__ __forceinline__ v16bf load_frag_contig(const uint16_t* p) {
  union { v4u q[2]; v16bf v; } u;
  u.q[0] = *reinterpret_cast<const v4u*>(p);
  u.q[1] = *reinterpret_cast<const v4u*>(p + 8);
  return u.v;
}

// async global->LDS 128-bit copy (CDNA5, tracked by ASYNCcnt; bypasses VGPRs)
__device__ __forceinline__ void async_copy_b128(uint32_t lds_off, const uint16_t* gptr) {
  asm volatile("global_load_async_to_lds_b128 %0, %1, off"
               :: "v"(lds_off), "v"(gptr) : "memory");
}

// ---------------- WMMA GEMM: C[M,N] = act(A[M,K] * BT[N,K]^T + bias) ----------------
// A row-major bf16, BT row-major bf16 (weights pre-transposed), C bf16.
// Block = 256 threads (8 wave32), tile 128x128, K-step 32.
// Triple-buffered async global->LDS pipeline, 2-tile lookahead, 1 barrier/tile.
#define BM   128
#define BN   128
#define BK   32
#define LDSS 40   // 32 k-elements + 8 pad (80B row stride, 16B aligned, bank-spread)

__device__ __forceinline__ void compute_tile(const uint16_t* __restrict__ pA,
                                             const uint16_t* __restrict__ pB,
                                             int wm, int wn, int lmod, int lhalf,
                                             v8f (&acc)[4][2]) {
  v16bf af[4], bfr[2];
#pragma unroll
  for (int i = 0; i < 4; ++i)         // A 16x32: lo lanes k{0..7,16..23}, hi k{8..15,24..31}
    af[i] = load_frag_split(pA + (wm * 64 + i * 16 + lmod) * LDSS + lhalf * 8);
#pragma unroll
  for (int j = 0; j < 2; ++j)         // B 32x16: lo lanes k{0..7,16..23}, hi k{8..15,24..31}
    bfr[j] = load_frag_split(pB + (wn * 32 + j * 16 + lmod) * LDSS + lhalf * 8);
#pragma unroll
  for (int i = 0; i < 4; ++i)
#pragma unroll
    for (int j = 0; j < 2; ++j)
      acc[i][j] = __builtin_amdgcn_wmma_f32_16x16x32_bf16(
          false, af[i], false, bfr[j], (short)0, acc[i][j], false, false);
}

template <int RELU>
__global__ __launch_bounds__(256) void gemm_bf16_rrt(
    const uint16_t* __restrict__ A,
    const uint16_t* __restrict__ BT,
    const float*    __restrict__ bias,
    uint16_t*       __restrict__ C,
    int M, int N, int K)
{
  __shared__ uint16_t As[3][BM * LDSS];
  __shared__ uint16_t Bs[3][BN * LDSS];

  const int tid   = threadIdx.x;
  const int lane  = tid & 31;
  const int wave  = tid >> 5;
  const int lmod  = lane & 15;
  const int lhalf = lane >> 4;          // 0: lanes 0-15, 1: lanes 16-31
  const int wm    = wave & 1;           // 2 waves along M (64 rows each)
  const int wn    = wave >> 1;          // 4 waves along N (32 cols each)
  const int bm0   = blockIdx.y * BM;
  const int bn0   = blockIdx.x * BN;

  const int srow = tid >> 1;            // staging row 0..127
  const int schk = (tid & 1) * 8;       // 16B chunk within 32-elem row

  // global source bases for this thread's staging slice
  const uint16_t* gA = A  + (size_t)(bm0 + srow) * K + schk;
  const uint16_t* gB = BT + (size_t)(bn0 + srow) * K + schk;

  // per-buffer staging LDS offsets (flat addr low 32 bits == LDS offset)
  uint32_t sA0 = (uint32_t)(uintptr_t)&As[0][srow * LDSS + schk];
  uint32_t sA1 = (uint32_t)(uintptr_t)&As[1][srow * LDSS + schk];
  uint32_t sA2 = (uint32_t)(uintptr_t)&As[2][srow * LDSS + schk];
  uint32_t sB0 = (uint32_t)(uintptr_t)&Bs[0][srow * LDSS + schk];
  uint32_t sB1 = (uint32_t)(uintptr_t)&Bs[1][srow * LDSS + schk];
  uint32_t sB2 = (uint32_t)(uintptr_t)&Bs[2][srow * LDSS + schk];
  // per-buffer fragment-load base pointers (rotated, never re-selected)
  const uint16_t* pA0 = &As[0][0];
  const uint16_t* pA1 = &As[1][0];
  const uint16_t* pA2 = &As[2][0];
  const uint16_t* pB0 = &Bs[0][0];
  const uint16_t* pB1 = &Bs[1][0];
  const uint16_t* pB2 = &Bs[2][0];

  v8f acc[4][2] = {};                   // 4 M-tiles x 2 N-tiles per wave

  // prologue: tiles 0 and 1 in flight (2 async ops per tile per wave)
  async_copy_b128(sA0, gA);
  async_copy_b128(sB0, gB);
  async_copy_b128(sA1, gA + BK);
  async_copy_b128(sB1, gB + BK);
  const uint16_t* gAn = gA + 2 * BK;    // next tile to fetch (tile 2)
  const uint16_t* gBn = gB + 2 * BK;

  const int nt = K / BK;                // >= 8 and even for all layers here

  // steady state: tiles 0 .. nt-3 (branch-free; unconditional prefetch)
  for (int kt = 0; kt < nt - 2; ++kt) {
    // retire tile kt: this wave has 4 ops outstanding (tiles kt, kt+1);
    // async loads complete in order, so <=2 means tile kt landed.
    asm volatile("s_wait_asynccnt 0x2" ::: "memory");
    __syncthreads();  // (a) tile kt resident for ALL waves
                      // (b) all waves finished compute(kt-1) -> slot2 reusable
    async_copy_b128(sA2, gAn);          // tile kt+2 -> slot computed at kt-1
    async_copy_b128(sB2, gBn);
    gAn += BK; gBn += BK;

    compute_tile(pA0, pB0, wm, wn, lmod, lhalf, acc);

    // rotate buffers: cur <- +1 <- +2 <- cur  (pure register moves)
    { uint32_t t = sA0; sA0 = sA1; sA1 = sA2; sA2 = t; }
    { uint32_t t = sB0; sB0 = sB1; sB1 = sB2; sB2 = t; }
    { const uint16_t* t = pA0; pA0 = pA1; pA1 = pA2; pA2 = t; }
    { const uint16_t* t = pB0; pB0 = pB1; pB1 = pB2; pB2 = t; }
  }

  // tail tile nt-2: tile nt-1 still in flight
  asm volatile("s_wait_asynccnt 0x2" ::: "memory");
  __syncthreads();
  compute_tile(pA0, pB0, wm, wn, lmod, lhalf, acc);
  { const uint16_t* t = pA0; pA0 = pA1; pA1 = pA2; pA2 = t; }
  { const uint16_t* t = pB0; pB0 = pB1; pB1 = pB2; pB2 = t; }

  // tail tile nt-1: drain
  asm volatile("s_wait_asynccnt 0x0" ::: "memory");
  __syncthreads();
  compute_tile(pA0, pB0, wm, wn, lmod, lhalf, acc);

  // epilogue: bias + compile-time ReLU, store bf16
#pragma unroll
  for (int i = 0; i < 4; ++i) {
    const int m0 = bm0 + wm * 64 + i * 16 + lhalf * 8;   // lanes 16-31 hold M rows +8
#pragma unroll
    for (int j = 0; j < 2; ++j) {
      const int n = bn0 + wn * 32 + j * 16 + lmod;
      const float bv = bias[n];
#pragma unroll
      for (int r = 0; r < 8; ++r) {
        float v = acc[i][j][r] + bv;
        if (RELU) v = fmaxf(v, 0.0f);
        C[(size_t)(m0 + r) * N + n] = f2bf(v);
      }
    }
  }
}

// ---------------- f32 -> bf16 cast ----------------
__global__ void cvt_f32_bf16(const float* __restrict__ in, uint16_t* __restrict__ out, int n) {
  int i = blockIdx.x * 256 + threadIdx.x;
  if (i < n) out[i] = f2bf(in[i]);
}

// batched [G,K,N] f32 -> [G,N,K] bf16 (weight transpose so GEMM reads contiguous k)
__global__ void tcvt_f32_bf16(const float* __restrict__ in, uint16_t* __restrict__ out,
                              int K, int N, int total) {
  int i = blockIdx.x * 256 + threadIdx.x;
  if (i >= total) return;
  int g = i / (K * N);
  int r = i - g * (K * N);
  int k = r / N;
  int n = r - k * N;
  out[(size_t)g * N * K + (size_t)n * K + k] = f2bf(in[i]);
}

// ---------------- gates: softmax_e(x @ gW + gb), one block per batch row ----------------
__global__ __launch_bounds__(256) void gate_softmax(
    const float* __restrict__ x, const float* __restrict__ gW,
    const float* __restrict__ gb, float* __restrict__ gates /*[B,T,E]*/)
{
  __shared__ float xs[D_];
  __shared__ float part[16][17];
  __shared__ float logits[16];
  const int tid = threadIdx.x;
  const int b   = blockIdx.x;
  for (int i = tid; i < D_; i += 256) xs[i] = x[(size_t)b * D_ + i];
  __syncthreads();
  const int j  = tid & 15;     // (t,e) combo
  const int ks = tid >> 4;     // 16-way k split
  const int t  = j >> 3, e = j & 7;
  float s = 0.0f;
  for (int k = ks; k < D_; k += 16)
    s += xs[k] * gW[((size_t)t * D_ + k) * E_ + e];
  part[j][ks] = s;
  __syncthreads();
  if (tid < 16) {
    float acc = 0.0f;
    for (int kk = 0; kk < 16; ++kk) acc += part[tid][kk];
    logits[tid] = acc + gb[(tid >> 3) * E_ + (tid & 7)];
  }
  __syncthreads();
  if (tid < T_) {
    float mx = -3.4e38f;
    for (int ee = 0; ee < E_; ++ee) mx = fmaxf(mx, logits[tid * E_ + ee]);
    float ex[E_], sum = 0.0f;
    for (int ee = 0; ee < E_; ++ee) { ex[ee] = __expf(logits[tid * E_ + ee] - mx); sum += ex[ee]; }
    float inv = 1.0f / sum;
    for (int ee = 0; ee < E_; ++ee)
      gates[((size_t)b * T_ + tid) * E_ + ee] = ex[ee] * inv;
  }
}

// ---------------- gate-weighted expert mix: tin[t,b,:] = sum_e g[b,t,e]*eo[e,b,:] ----------------
__global__ void mix_experts(const float* __restrict__ gates,
                            const uint16_t* __restrict__ eo,   // [E,B,DO]
                            uint16_t* __restrict__ tin)        // [T,B,DO]
{
  const int d = threadIdx.x;   // blockDim = DO = 256
  const int b = blockIdx.x;
  float v0 = 0.0f, v1 = 0.0f;
#pragma unroll
  for (int e = 0; e < E_; ++e) {
    float ev = bf2f(eo[((size_t)e * B_ + b) * DO_ + d]);
    v0 += gates[(size_t)b * (T_ * E_) + e]      * ev;
    v1 += gates[(size_t)b * (T_ * E_) + E_ + e] * ev;
  }
  tin[(size_t)b * DO_ + d]                = f2bf(v0);
  tin[((size_t)B_ + (size_t)b) * DO_ + d] = f2bf(v1);
}

// ---------------- tower head: out[b,t] = sigmoid(th2[b,:] . w + bias), wave per row ----------------
__global__ __launch_bounds__(256) void tower_head(
    const uint16_t* __restrict__ th2,  // [B,TH2] bf16
    const float* __restrict__ w,       // [TH2]
    const float* __restrict__ bptr,    // scalar
    float* __restrict__ out, int t)
{
  const int lane = threadIdx.x & 31;
  const int wv   = threadIdx.x >> 5;
  const int b    = blockIdx.x * 8 + wv;
  float s = 0.0f;
#pragma unroll
  for (int d = lane; d < TH2_; d += 32)
    s += bf2f(th2[(size_t)b * TH2_ + d]) * w[d];
#pragma unroll
  for (int off = 16; off > 0; off >>= 1)
    s += __shfl_down(s, off, 32);
  if (lane == 0)
    out[(size_t)b * T_ + t] = 1.0f / (1.0f + __expf(-(s + bptr[0])));
}

// ---------------- host orchestration ----------------
extern "C" void kernel_launch(void* const* d_in, const int* in_sizes, int n_in,
                              void* d_out, int out_size, void* d_ws, size_t ws_size,
                              hipStream_t stream) {
  (void)in_sizes; (void)n_in; (void)out_size; (void)ws_size;
  const float* x   = (const float*)d_in[0];
  const float* eW1 = (const float*)d_in[1];
  const float* eb1 = (const float*)d_in[2];
  const float* eW2 = (const float*)d_in[3];
  const float* eb2 = (const float*)d_in[4];
  const float* eW3 = (const float*)d_in[5];
  const float* eb3 = (const float*)d_in[6];
  const float* gW  = (const float*)d_in[7];
  const float* gb  = (const float*)d_in[8];
  const float* tW1 = (const float*)d_in[9];
  const float* tb1 = (const float*)d_in[10];
  const float* tW2 = (const float*)d_in[11];
  const float* tb2 = (const float*)d_in[12];
  const float* tW3 = (const float*)d_in[13];
  const float* tb3 = (const float*)d_in[14];
  float* out = (float*)d_out;

  // workspace carve-up (~220 MB total; h1/h2/th1/th2 reused across experts/tasks)
  char* ws = (char*)d_ws;
  size_t off = 0;
  auto take = [&](size_t bytes) -> char* {
    char* p = ws + off;
    off += (bytes + 255) & ~(size_t)255;
    return p;
  };
  uint16_t* xb    = (uint16_t*)take((size_t)B_ * D_ * 2);
  uint16_t* w1t   = (uint16_t*)take((size_t)E_ * H1_ * D_ * 2);
  uint16_t* w2t   = (uint16_t*)take((size_t)E_ * H2_ * H1_ * 2);
  uint16_t* w3t   = (uint16_t*)take((size_t)E_ * DO_ * H2_ * 2);
  uint16_t* tw1t  = (uint16_t*)take((size_t)T_ * TH1_ * DO_ * 2);
  uint16_t* tw2t  = (uint16_t*)take((size_t)T_ * TH2_ * TH1_ * 2);
  uint16_t* h1    = (uint16_t*)take((size_t)B_ * H1_ * 2);
  uint16_t* h2    = (uint16_t*)take((size_t)B_ * H2_ * 2);
  uint16_t* eo    = (uint16_t*)take((size_t)E_ * B_ * DO_ * 2);
  float*    gates = (float*)   take((size_t)B_ * T_ * E_ * 4);
  uint16_t* tin   = (uint16_t*)take((size_t)T_ * B_ * DO_ * 2);
  uint16_t* th1   = (uint16_t*)take((size_t)B_ * TH1_ * 2);
  uint16_t* th2   = (uint16_t*)take((size_t)B_ * TH2_ * 2);

  // precision casts + weight transposes ([K,N] -> [N,K] bf16)
  cvt_f32_bf16 <<<(B_*D_)/256,        256, 0, stream>>>(x,   xb,   B_*D_);
  tcvt_f32_bf16<<<(E_*D_*H1_)/256,    256, 0, stream>>>(eW1, w1t,  D_,   H1_,  E_*D_*H1_);
  tcvt_f32_bf16<<<(E_*H1_*H2_)/256,   256, 0, stream>>>(eW2, w2t,  H1_,  H2_,  E_*H1_*H2_);
  tcvt_f32_bf16<<<(E_*H2_*DO_)/256,   256, 0, stream>>>(eW3, w3t,  H2_,  DO_,  E_*H2_*DO_);
  tcvt_f32_bf16<<<(T_*DO_*TH1_)/256,  256, 0, stream>>>(tW1, tw1t, DO_,  TH1_, T_*DO_*TH1_);
  tcvt_f32_bf16<<<(T_*TH1_*TH2_)/256, 256, 0, stream>>>(tW2, tw2t, TH1_, TH2_, T_*TH1_*TH2_);

  gate_softmax<<<B_, 256, 0, stream>>>(x, gW, gb, gates);

  auto gemm = [&](const uint16_t* A, const uint16_t* BT, const float* bias,
                  uint16_t* C, int M, int N, int K, int relu) {
    dim3 g(N / BN, M / BM);
    if (relu) gemm_bf16_rrt<1><<<g, 256, 0, stream>>>(A, BT, bias, C, M, N, K);
    else      gemm_bf16_rrt<0><<<g, 256, 0, stream>>>(A, BT, bias, C, M, N, K);
  };

  // experts (sequential per expert; h1/h2 reused, eo holds [E,B,DO])
  for (int e = 0; e < E_; ++e) {
    gemm(xb, w1t + (size_t)e * H1_ * D_,  eb1 + (size_t)e * H1_, h1,                        B_, H1_, D_,  1);
    gemm(h1, w2t + (size_t)e * H2_ * H1_, eb2 + (size_t)e * H2_, h2,                        B_, H2_, H1_, 1);
    gemm(h2, w3t + (size_t)e * DO_ * H2_, eb3 + (size_t)e * DO_, eo + (size_t)e * B_ * DO_, B_, DO_, H2_, 0);
  }

  mix_experts<<<B_, 256, 0, stream>>>(gates, eo, tin);

  // towers (sequential per task; th1/th2 reused)
  for (int t = 0; t < T_; ++t) {
    gemm(tin + (size_t)t * B_ * DO_, tw1t + (size_t)t * TH1_ * DO_,  tb1 + (size_t)t * TH1_, th1, B_, TH1_, DO_,  1);
    gemm(th1,                        tw2t + (size_t)t * TH2_ * TH1_, tb2 + (size_t)t * TH2_, th2, B_, TH2_, TH1_, 1);
    tower_head<<<B_/8, 256, 0, stream>>>(th2, tW3 + (size_t)t * TH2_, tb3 + t, out, t);
  }
}
